// MAGClassifier_3496103379492
// MI455X (gfx1250) — matrix operand order repaired and checked by reading, not compiled, not run on hardware
//
#include <hip/hip_runtime.h>
#include <hip/hip_bf16.h>

typedef __attribute__((ext_vector_type(16))) _Float16 v16h;
typedef __attribute__((ext_vector_type(8)))  _Float16 v8h;
typedef __attribute__((ext_vector_type(8)))  float    v8f;
typedef __attribute__((ext_vector_type(4)))  float    v4f;

#define NEDGE 4096
#define DMODEL 256
#define NHEAD 8
#define DHEAD 32

// ---------------------------------------------------------------------------
// A-fragment loader: fp32 row -> f16 A layout (16x32, lane row = l16).
// halves h=0..7  -> K = h + 8*half        (contiguous run of 8 floats)
// halves h=8..15 -> K = 16 + (h-8) + 8*half (second contiguous run)
// ---------------------------------------------------------------------------
__device__ inline v16h load_a_frag(const float* __restrict__ Arow, int half)
{
    const v4f* p = (const v4f*)Arow;
    v4f a0 = p[half * 2 + 0];
    v4f a1 = p[half * 2 + 1];
    v4f a2 = p[4 + half * 2 + 0];
    v4f a3 = p[4 + half * 2 + 1];
    v16h a;
    #pragma unroll
    for (int i = 0; i < 4; ++i) {
        a[i]      = (_Float16)a0[i];
        a[4 + i]  = (_Float16)a1[i];
        a[8 + i]  = (_Float16)a2[i];
        a[12 + i] = (_Float16)a3[i];
    }
    return a;
}

// ---------------------------------------------------------------------------
// WMMA GEMM, one wave per 16(M) x 64(N) tile, K stepped by 32.
// A loaded once per K-step (4x float4), reused by 4 WMMAs.
// MODE 0: Cf = A@B + bias            (fp32 out)
// MODE 1: Ch = f16(A@B + bias)       (f16 out, feeds attention)
// MODE 2: Cf = res + relu(A@B+bias)  (MAB output path)
// ---------------------------------------------------------------------------
template<int MODE>
__global__ __launch_bounds__(32)
void gemm16(const float* __restrict__ A, const float* __restrict__ B,
            const float* __restrict__ bias, const float* __restrict__ res,
            float* __restrict__ Cf, _Float16* __restrict__ Ch,
            int M, int N, int K)
{
    const int tiles_n = N >> 6;            // 64-wide N tiles
    const int tm = blockIdx.x / tiles_n;
    const int tn = blockIdx.x % tiles_n;
    const int lane = threadIdx.x;
    const int half = lane >> 4;
    const int l16  = lane & 15;
    const int row  = tm * 16 + l16;

    v8f c0 = {}, c1 = {}, c2 = {}, c3 = {};
    for (int k0 = 0; k0 < K; k0 += 32) {
        const float* Arow = A + (size_t)row * K + k0;
        v16h a = load_a_frag(Arow, half);

        v16h b0, b1, b2, b3;
        #pragma unroll
        for (int h = 0; h < 16; ++h) {
            // B 32x16 layout: lane col = l16, K = (h<8?h:h+8) + 8*half
            const float* Brow = B + (size_t)(k0 + (h < 8 ? h : h + 8) + 8 * half) * N + tn * 64 + l16;
            b0[h] = (_Float16)Brow[0];
            b1[h] = (_Float16)Brow[16];
            b2[h] = (_Float16)Brow[32];
            b3[h] = (_Float16)Brow[48];
        }
        c0 = __builtin_amdgcn_wmma_f32_16x16x32_f16(false, a, false, b0, (short)0, c0, false, false);
        c1 = __builtin_amdgcn_wmma_f32_16x16x32_f16(false, a, false, b1, (short)0, c1, false, false);
        c2 = __builtin_amdgcn_wmma_f32_16x16x32_f16(false, a, false, b2, (short)0, c2, false, false);
        c3 = __builtin_amdgcn_wmma_f32_16x16x32_f16(false, a, false, b3, (short)0, c3, false, false);
    }

    #pragma unroll
    for (int sub = 0; sub < 4; ++sub) {
        const v8f& c = (sub == 0) ? c0 : (sub == 1) ? c1 : (sub == 2) ? c2 : c3;
        const int col = tn * 64 + sub * 16 + l16;
        const float bv = bias[col];
        #pragma unroll
        for (int r = 0; r < 8; ++r) {
            int m = r + half * 8;                    // C layout: row M = r + 8*half
            size_t idx = (size_t)(tm * 16 + m) * N + col;
            float val = c[r] + bv;
            if constexpr (MODE == 0) {
                Cf[idx] = val;
            } else if constexpr (MODE == 1) {
                Ch[idx] = (_Float16)val;
            } else {
                Cf[idx] = res[idx] + fmaxf(val, 0.0f);
            }
        }
    }
}

// ---------------------------------------------------------------------------
// er[e] = concat(node[src[e]], node[dst[e]]) + concat(ef[e], ef[e])
// ---------------------------------------------------------------------------
__global__ void build_er(const float* __restrict__ node, const float* __restrict__ ef,
                         const int* __restrict__ src, const int* __restrict__ dst,
                         float* __restrict__ er)
{
    int i = blockIdx.x * blockDim.x + threadIdx.x;
    if (i >= NEDGE * DMODEL) return;
    int e = i >> 8, c = i & 255;
    float val;
    if (c < 128) val = node[(size_t)src[e] * 128 + c]       + ef[(size_t)e * 128 + c];
    else         val = node[(size_t)dst[e] * 128 + (c-128)] + ef[(size_t)e * 128 + (c-128)];
    er[i] = val;
}

// ---------------------------------------------------------------------------
// Masked flash attention over f16 q/k/v, one wave per (16-query block, head).
//   S^T = K_h * Q_h^T   (output col = query -> lane-local flash state)
//   O^T = V_h^T * P^T   accumulated in fp32
//   o[q, h*32+d] = q[q, h*32+d] + (softmax(S) @ v)[q, d]
// ---------------------------------------------------------------------------
__global__ __launch_bounds__(32)
void attn16(const _Float16* __restrict__ qm, const _Float16* __restrict__ km,
            const _Float16* __restrict__ vm, const int* __restrict__ src,
            const int* __restrict__ dst, float* __restrict__ om)
{
    const int head = blockIdx.y;
    const int qb   = blockIdx.x;
    const int lane = threadIdx.x;
    const int half = lane >> 4;
    const int l16  = lane & 15;
    const int hoff = head * DHEAD;
    const int qi   = qb * 16 + l16;          // query this lane's column owns

    // B_q: 32(dh) x 16(q): lane col = l16, K(dh) = h + 16*half -> contiguous 16 halves
    const _Float16* qrow = qm + (size_t)qi * DMODEL + hoff;
    v16h bq = *(const v16h*)(qrow + 16 * half);

    const int sq = src[qi], dq = dst[qi];

    float m = -3.0e38f, l = 0.0f;
    v8f ot0 = {}, ot1 = {};                  // O^T tiles: dh 0-15 / dh 16-31

    const _Float16* k0p = km + (size_t)l16 * DMODEL + hoff;

    for (int kb = 0; kb < NEDGE; kb += 32) {
        const _Float16* kp0 = k0p + (size_t)kb * DMODEL;          // keys kb..kb+15
        const _Float16* kp1 = kp0 + (size_t)16 * DMODEL;          // keys kb+16..kb+31

        // prefetch next key-block rows of K and V (emits global_prefetch_b8)
        if (kb + 32 < NEDGE) {
            __builtin_prefetch((const void*)(kp0 + (size_t)32 * DMODEL), 0, 0);
            __builtin_prefetch((const void*)(vm + (size_t)(kb + 32 + l16) * DMODEL + hoff), 0, 0);
        }

        // A_k: rows = keys (lane row l16), K(dh) = h + 16*half
        //      -> two contiguous 8-half runs at +16*half and +16*half+8
        v8h a0lo = *(const v8h*)(kp0 + half * 16);
        v8h a0hi = *(const v8h*)(kp0 + half * 16 + 8);
        v8h a1lo = *(const v8h*)(kp1 + half * 16);
        v8h a1hi = *(const v8h*)(kp1 + half * 16 + 8);
        v16h ak0 = __builtin_shufflevector(a0lo, a0hi, 0,1,2,3,4,5,6,7,8,9,10,11,12,13,14,15);
        v16h ak1 = __builtin_shufflevector(a1lo, a1hi, 0,1,2,3,4,5,6,7,8,9,10,11,12,13,14,15);

        v8f z = {};
        v8f s0 = __builtin_amdgcn_wmma_f32_16x16x32_f16(false, ak0, false, bq, (short)0, z, false, false);
        v8f s1 = __builtin_amdgcn_wmma_f32_16x16x32_f16(false, ak1, false, bq, (short)0, z, false, false);

        // scale 1/sqrt(256), mask, block max
        float p0[8], p1[8];
        float bmax = -3.0e38f;
        #pragma unroll
        for (int r = 0; r < 8; ++r) {
            int key0 = kb + r + half * 8;        // tile0 row -> key index
            int key1 = key0 + 16;                // tile1 row
            int sk0 = src[key0], dk0 = dst[key0];
            int sk1 = src[key1], dk1 = dst[key1];
            bool m0 = (sq == sk0) | (sq == dk0) | (dq == sk0) | (dq == dk0);
            bool m1 = (sq == sk1) | (sq == dk1) | (dq == sk1) | (dq == dk1);
            float v0 = m0 ? s0[r] * 0.0625f : -1.0e9f;
            float v1 = m1 ? s1[r] * 0.0625f : -1.0e9f;
            p0[r] = v0; p1[r] = v1;
            bmax = fmaxf(bmax, fmaxf(v0, v1));
        }
        // each query column lives in lanes L and L^16: merge halves
        bmax = fmaxf(bmax, __shfl_xor(bmax, 16, 32));
        float mnew = fmaxf(m, bmax);
        float resc = __expf(m - mnew);
        float bsum = 0.0f;
        #pragma unroll
        for (int r = 0; r < 8; ++r) {
            p0[r] = __expf(p0[r] - mnew);
            p1[r] = __expf(p1[r] - mnew);
            bsum += p0[r] + p1[r];
        }
        bsum += __shfl_xor(bsum, 16, 32);
        l = l * resc + bsum;
        #pragma unroll
        for (int r = 0; r < 8; ++r) { ot0[r] *= resc; ot1[r] *= resc; }
        m = mnew;

        // B_p (32 keys x 16 q): lane col=q, K(keys) = h + 16*half.
        // lanes<16: keys 0-7 own p0, keys 8-15 partner p0;
        // lanes>=16: keys 16-23 partner p1, keys 24-31 own p1.
        v16h bp;
        #pragma unroll
        for (int r = 0; r < 8; ++r) {
            float r0 = __shfl_xor(p0[r], 16, 32);
            float r1 = __shfl_xor(p1[r], 16, 32);
            bp[r]     = (_Float16)(half ? r1    : p0[r]);
            bp[r + 8] = (_Float16)(half ? p1[r] : r0);
        }

        // A_v: V^T, rows = dh (lane row l16 / l16+16), K(keys)=h+16*half
        v16h av0, av1;
        #pragma unroll
        for (int h = 0; h < 16; ++h) {
            int key = kb + h + half * 16;
            const _Float16* vr = vm + (size_t)key * DMODEL + hoff;
            av0[h] = vr[l16];
            av1[h] = vr[16 + l16];
        }
        ot0 = __builtin_amdgcn_wmma_f32_16x16x32_f16(false, av0, false, bp, (short)0, ot0, false, false);
        ot1 = __builtin_amdgcn_wmma_f32_16x16x32_f16(false, av1, false, bp, (short)0, ot1, false, false);
    }

    // epilogue: o[q][h*32+dh] = q[q][h*32+dh] + O^T[dh][q] / l
    const float inv = 1.0f / l;
    const size_t base = (size_t)qi * DMODEL + hoff;
    #pragma unroll
    for (int r = 0; r < 8; ++r) {
        int dh0 = r + half * 8;
        om[base + dh0]      = (float)qm[base + dh0]      + ot0[r] * inv;
        om[base + dh0 + 16] = (float)qm[base + dh0 + 16] + ot1[r] * inv;
    }
}

// ---------------------------------------------------------------------------
// Single-workgroup finalize: PMA query proj, per-head pooling attention
// (one wave = one head), PMA output proj, cls1, cls2 -> scalar.
// ---------------------------------------------------------------------------
__global__ __launch_bounds__(256)
void finalize(const float* __restrict__ seed,
              const float* __restrict__ wq, const float* __restrict__ bq,
              const float* __restrict__ kp, const float* __restrict__ vp,
              const float* __restrict__ wo, const float* __restrict__ bo,
              const float* __restrict__ w1, const float* __restrict__ b1,
              const float* __restrict__ w2, const float* __restrict__ b2,
              float* __restrict__ out)
{
    __shared__ float qg[DMODEL], g[DMODEL], g2[DMODEL], hid[128], red[8];
    const int t = threadIdx.x;

    // q_pma = seed @ Wq + bq
    {
        float acc = bq[t];
        for (int i = 0; i < DMODEL; ++i) acc += seed[i] * wq[(size_t)i * DMODEL + t];
        qg[t] = acc;
    }
    __syncthreads();

    // attention pooling: wave w handles head w (32 lanes stream 4096 keys)
    const int head = t >> 5, lane = t & 31, hoff = head * DHEAD;
    float qh[DHEAD];
    #pragma unroll
    for (int d = 0; d < DHEAD; ++d) qh[d] = qg[hoff + d];

    float m = -3.0e38f, l = 0.0f, accv[DHEAD];
    #pragma unroll
    for (int d = 0; d < DHEAD; ++d) accv[d] = 0.0f;

    for (int mi = lane; mi < NEDGE; mi += 32) {
        const float* kr = kp + (size_t)mi * DMODEL + hoff;
        float s = 0.0f;
        #pragma unroll
        for (int d = 0; d < DHEAD; ++d) s += qh[d] * kr[d];
        s *= 0.0625f;
        float mn = fmaxf(m, s);
        float sc = __expf(m - mn);
        float p  = __expf(s - mn);
        l = l * sc + p;
        const float* vr = vp + (size_t)mi * DMODEL + hoff;
        #pragma unroll
        for (int d = 0; d < DHEAD; ++d) accv[d] = accv[d] * sc + p * vr[d];
        m = mn;
    }
    // wave-level flash-state combine
    for (int off = 1; off < 32; off <<= 1) {
        float mo = __shfl_xor(m, off, 32);
        float lo = __shfl_xor(l, off, 32);
        float mn = fmaxf(m, mo);
        float sa = __expf(m - mn), sb = __expf(mo - mn);
        l = l * sa + lo * sb;
        #pragma unroll
        for (int d = 0; d < DHEAD; ++d) {
            float ao = __shfl_xor(accv[d], off, 32);
            accv[d] = accv[d] * sa + ao * sb;
        }
        m = mn;
    }
    g[hoff + lane] = qg[hoff + lane] + accv[lane] / l;
    __syncthreads();

    // PMA output: g2 = g + relu(g @ Wo + bo)
    {
        float acc = bo[t];
        for (int i = 0; i < DMODEL; ++i) acc += g[i] * wo[(size_t)i * DMODEL + t];
        g2[t] = g[t] + fmaxf(acc, 0.0f);
    }
    __syncthreads();

    // cls1: hid = relu(g2 @ W1 + b1)  [256 -> 128]
    if (t < 128) {
        float acc = b1[t];
        for (int i = 0; i < DMODEL; ++i) acc += g2[i] * w1[(size_t)i * 128 + t];
        hid[t] = fmaxf(acc, 0.0f);
    }
    __syncthreads();

    // cls2: out = hid @ W2 + b2  [128 -> 1]
    float a4 = (t < 128) ? hid[t] * w2[t] : 0.0f;
    for (int off = 16; off > 0; off >>= 1) a4 += __shfl_xor(a4, off, 32);
    if (lane == 0) red[head] = a4;
    __syncthreads();
    if (t == 0) {
        float s = b2[0];
        for (int w = 0; w < 8; ++w) s += red[w];
        out[0] = s;
    }
}

// ---------------------------------------------------------------------------
extern "C" void kernel_launch(void* const* d_in, const int* in_sizes, int n_in,
                              void* d_out, int out_size, void* d_ws, size_t ws_size,
                              hipStream_t stream)
{
    (void)in_sizes; (void)n_in; (void)out_size; (void)ws_size;

    const float* x   = (const float*)d_in[0];
    const int*   ei  = (const int*)  d_in[1];
    const float* ea  = (const float*)d_in[2];
    // params flattened in dict insertion order, each linear = (w, b)
    const float* nw  = (const float*)d_in[3];
    const float* nb  = (const float*)d_in[4];
    const float* ew  = (const float*)d_in[5];
    const float* eb  = (const float*)d_in[6];
    const float* s1qw=(const float*)d_in[7],  *s1qb=(const float*)d_in[8];
    const float* s1kw=(const float*)d_in[9],  *s1kb=(const float*)d_in[10];
    const float* s1vw=(const float*)d_in[11], *s1vb=(const float*)d_in[12];
    const float* s1ow=(const float*)d_in[13], *s1ob=(const float*)d_in[14];
    const float* s2qw=(const float*)d_in[15], *s2qb=(const float*)d_in[16];
    const float* s2kw=(const float*)d_in[17], *s2kb=(const float*)d_in[18];
    const float* s2vw=(const float*)d_in[19], *s2vb=(const float*)d_in[20];
    const float* s2ow=(const float*)d_in[21], *s2ob=(const float*)d_in[22];
    const float* pqw =(const float*)d_in[23], *pqb =(const float*)d_in[24];
    const float* pkw =(const float*)d_in[25], *pkb =(const float*)d_in[26];
    const float* pvw =(const float*)d_in[27], *pvb =(const float*)d_in[28];
    const float* pow_=(const float*)d_in[29], *pob =(const float*)d_in[30];
    const float* seed=(const float*)d_in[31];
    const float* c1w =(const float*)d_in[32], *c1b =(const float*)d_in[33];
    const float* c2w =(const float*)d_in[34], *c2b =(const float*)d_in[35];

    const int* src = ei;
    const int* dst = ei + NEDGE;

    // workspace layout (float units)
    float*      ws   = (float*)d_ws;
    float*      node = ws;                         // 1024*128
    float*      ef   = node + 1024 * 128;          // 4096*128
    float*      hA   = ef   + 4096 * 128;          // 4096*256 fp32
    float*      hB   = hA   + NEDGE * DMODEL;      // 4096*256 fp32
    float*      ob_  = hB   + NEDGE * DMODEL;      // 4096*256 fp32 (attn out / PMA-K)
    _Float16*   qh16 = (_Float16*)(ob_ + NEDGE * DMODEL);   // 4096*256 f16
    _Float16*   kh16 = qh16 + (size_t)NEDGE * DMODEL;
    _Float16*   vh16 = kh16 + (size_t)NEDGE * DMODEL;

    const int gt_h   = (NEDGE/16)*(128/64);        // [4096,128] tiles
    const int gt_d   = (NEDGE/16)*(DMODEL/64);     // [4096,256] tiles
    dim3 agrid(NEDGE/16, NHEAD);

    // encoders (fp32 out)
    gemm16<0><<<(1024/16)*(128/64), 32, 0, stream>>>(x,  nw, nb, nullptr, node, nullptr, 1024, 128, 64);
    gemm16<0><<<gt_h, 32, 0, stream>>>(ea, ew, eb, nullptr, ef, nullptr, NEDGE, 128, 32);
    build_er<<<(NEDGE*DMODEL)/256, 256, 0, stream>>>(node, ef, src, dst, hA);

    // SAB1: hA -> hB   (q/k/v in f16 for the matrix engine)
    gemm16<1><<<gt_d, 32, 0, stream>>>(hA, s1qw, s1qb, nullptr, nullptr, qh16, NEDGE, DMODEL, DMODEL);
    gemm16<1><<<gt_d, 32, 0, stream>>>(hA, s1kw, s1kb, nullptr, nullptr, kh16, NEDGE, DMODEL, DMODEL);
    gemm16<1><<<gt_d, 32, 0, stream>>>(hA, s1vw, s1vb, nullptr, nullptr, vh16, NEDGE, DMODEL, DMODEL);
    attn16<<<agrid, 32, 0, stream>>>(qh16, kh16, vh16, src, dst, ob_);
    gemm16<2><<<gt_d, 32, 0, stream>>>(ob_, s1ow, s1ob, ob_, hB, nullptr, NEDGE, DMODEL, DMODEL);

    // SAB2: hB -> hA
    gemm16<1><<<gt_d, 32, 0, stream>>>(hB, s2qw, s2qb, nullptr, nullptr, qh16, NEDGE, DMODEL, DMODEL);
    gemm16<1><<<gt_d, 32, 0, stream>>>(hB, s2kw, s2kb, nullptr, nullptr, kh16, NEDGE, DMODEL, DMODEL);
    gemm16<1><<<gt_d, 32, 0, stream>>>(hB, s2vw, s2vb, nullptr, nullptr, vh16, NEDGE, DMODEL, DMODEL);
    attn16<<<agrid, 32, 0, stream>>>(qh16, kh16, vh16, src, dst, ob_);
    gemm16<2><<<gt_d, 32, 0, stream>>>(ob_, s2ow, s2ob, ob_, hA, nullptr, NEDGE, DMODEL, DMODEL);

    // PMA projections over hA (fp32 out; ob_ and hB are free now)
    gemm16<0><<<gt_d, 32, 0, stream>>>(hA, pkw, pkb, nullptr, ob_, nullptr, NEDGE, DMODEL, DMODEL);
    gemm16<0><<<gt_d, 32, 0, stream>>>(hA, pvw, pvb, nullptr, hB,  nullptr, NEDGE, DMODEL, DMODEL);

    // PMA attention + classifier head -> scalar
    finalize<<<1, 256, 0, stream>>>(seed, pqw, pqb, ob_, hB, pow_, pob,
                                    c1w, c1b, c2w, c2b, (float*)d_out);
}
